// SynthesisLayer_43104291782964
// MI455X (gfx1250) — compile-verified, matrix operand order checked by reading, not run on hardware
//
#include <hip/hip_runtime.h>
#include <cstdint>

// ---------------------------------------------------------------------------
// StyleGAN modulated 3D conv for MI455X (gfx1250).
//   B=4, Cin=Cout=128, RES=32, K=3, W_DIM=512, f32 in/out.
// Strategy: implicit GEMM on v_wmma_f32_16x16x32_bf16 (bf16 data, f32 accum),
// 27 shifted 128x128 GEMM taps. x halo staged once in LDS; per-tap weight
// tiles double-buffered in LDS via gfx1250 async global->LDS loads (ASYNCcnt)
// so the 32KB weight fetch overlaps WMMA compute.
// ---------------------------------------------------------------------------

typedef __bf16 bf16;
typedef __attribute__((ext_vector_type(16))) __bf16        v16bf;
typedef __attribute__((ext_vector_type(8)))  float         v8f;
typedef __attribute__((ext_vector_type(4)))  unsigned int  u32x4;
typedef __attribute__((ext_vector_type(4)))  int           v4i;

#define CIN   128
#define COUT  128
#define RES   32
#define DHW   (RES * RES * RES)
#define WDIM  512
#define NTAP  27
#define NPOS  (9 * 34)   // 3x3 (d,h) halo rows x 34 padded w positions
#define XSTR  136        // 128 ci + 8 pad (272B rows -> bank stride 4 dwords)
#define ASTR  136

#define AS1 __attribute__((address_space(1)))
#define AS3 __attribute__((address_space(3)))

#if __has_builtin(__builtin_amdgcn_global_load_async_to_lds_b128)
#define USE_ASYNC 1
#else
#define USE_ASYNC 0
#endif

// ---------------------------------------------------------------------------
// Kernel 1: styles[b][ci] = (w[b] . affine_weight[ci]) / sqrt(512) + bias[ci]
// ---------------------------------------------------------------------------
__global__ void styles_kernel(const float* __restrict__ w,
                              const float* __restrict__ aw,
                              const float* __restrict__ ab,
                              float* __restrict__ styles) {
    int idx = blockIdx.x * blockDim.x + threadIdx.x;
    if (idx >= 4 * CIN) return;
    int b = idx >> 7, ci = idx & 127;
    const float* wr = w + b * WDIM;
    const float* ar = aw + (size_t)ci * WDIM;
    float acc = 0.f;
    for (int k = 0; k < WDIM; ++k) acc += wr[k] * ar[k];
    styles[idx] = acc * 0.044194173824159216f /* 1/sqrt(512) */ + ab[ci];
}

// ---------------------------------------------------------------------------
// Kernel 2: modulate + demodulate, emit bf16 weights as [b][tap][co][ci]
// (per-tap A tile = contiguous 128x128 bf16 = 32KB)
// ---------------------------------------------------------------------------
__global__ __launch_bounds__(128) void modulate_kernel(
        const float* __restrict__ weight,   // [Cout][Cin][27]
        const float* __restrict__ styles,   // [B][Cin]
        bf16* __restrict__ wmod) {          // [B][27][Cout][Cin]
    int b  = blockIdx.x >> 7;
    int co = blockIdx.x & 127;
    int ci = threadIdx.x;

    float s = styles[b * CIN + ci];
    const float* wp = weight + ((size_t)co * CIN + ci) * NTAP;
    float wl[NTAP];
    float sq = 0.f;
#pragma unroll
    for (int t = 0; t < NTAP; ++t) {
        float v = wp[t] * s;
        wl[t] = v;
        sq += v * v;
    }
    __shared__ float red[CIN];
    red[ci] = sq;
    __syncthreads();
    for (int off = 64; off > 0; off >>= 1) {
        if (ci < off) red[ci] += red[ci + off];
        __syncthreads();
    }
    float dcoef = rsqrtf(red[0] + 1e-8f);

    bf16* o = wmod + (size_t)b * NTAP * COUT * CIN + (size_t)co * CIN + ci;
#pragma unroll
    for (int t = 0; t < NTAP; ++t)
        o[(size_t)t * COUT * CIN] = (bf16)(wl[t] * dcoef);
}

// ---------------------------------------------------------------------------
// Fragment loader: 16 bf16 per lane = two ds_load_b128 (32B apart per ISA
// 16-bit A/B layout: lanes 0-15 hold K 0..7 & 16..23, lanes 16-31 the rest).
// ---------------------------------------------------------------------------
__device__ __forceinline__ v16bf load_frag(const bf16* p) {
    union { u32x4 q[2]; v16bf v; } u;
    const u32x4* s = (const u32x4*)p;
    u.q[0] = s[0];   // K 0..7   (relative)
    u.q[1] = s[2];   // K 16..23 (relative, +32B)
    return u.v;
}

// ---------------------------------------------------------------------------
// Stage one 128x128 bf16 weight tile (32KB) global -> LDS.
// Async path: 8 x global_load_async_to_lds_b128 per thread (ASYNCcnt),
// no VGPR round-trip. Fallback: b128 load + ds_store.
// ---------------------------------------------------------------------------
__device__ __forceinline__ void stage_a_tile(const bf16* g, bf16* l, int tid) {
#if USE_ASYNC
    for (int i = tid; i < 2048; i += 256) {          // 2048 x 16B
        int co = i >> 4, q = i & 15;
        __builtin_amdgcn_global_load_async_to_lds_b128(
            (AS1 v4i*)(uintptr_t)(g + (size_t)co * CIN + q * 8),
            (AS3 v4i*)(uintptr_t)(l + co * ASTR + q * 8),
            0, 0);
    }
#else
    for (int i = tid; i < 2048; i += 256) {
        int co = i >> 4, q = i & 15;
        *(u32x4*)&l[co * ASTR + q * 8] =
            *(const u32x4*)(g + (size_t)co * CIN + q * 8);
    }
    __builtin_prefetch(g + (size_t)COUT * CIN + tid * 64, 0, 1);
#endif
}

// ---------------------------------------------------------------------------
// Kernel 3: main conv. One block per (b, d, h): 128 Cout x 32 w positions.
// 8 waves; wave owns 16 Cout rows x two 16-wide N tiles -> 2 x v8f accum.
// ---------------------------------------------------------------------------
__global__ __launch_bounds__(256, 1) void conv_kernel(
        const float* __restrict__ x,      // [B][Cin][D][H][W]
        const bf16*  __restrict__ wmod,   // [B][27][Cout][Cin]
        const float* __restrict__ noise,  // [D][H][W]
        const float* __restrict__ nstr,   // [1]
        const float* __restrict__ bias,   // [Cout]
        float* __restrict__ out) {        // [B][Cout][D][H][W]
    __shared__ __align__(16) bf16 x_lds[NPOS * XSTR];      // 83,232 B
    __shared__ __align__(16) bf16 a_lds[2][COUT * ASTR];   // 2 x 34,816 B

    const int tid = threadIdx.x;
    const int b   = blockIdx.x >> 10;
    const int dh  = blockIdx.x & 1023;
    const int d   = dh >> 5, h = dh & 31;

    const bf16* wsrc = wmod + (size_t)b * NTAP * COUT * CIN;

    // Prologue: kick off tile 0 (async path overlaps with x staging below)
    stage_a_tile(wsrc, a_lds[0], tid);

    // ---- Stage x halo tile: 9 (d,h) rows x 34 w x 128 ci, f32 -> bf16 ----
    const float* xb = x + (size_t)b * CIN * DHW;
    for (int idx = tid; idx < 306 * CIN; idx += 256) {
        int ci   = idx / 306;           // 306 = 9*34
        int rem  = idx - ci * 306;
        int row  = rem / 34;            // dz*3 + dy
        int wcol = rem - row * 34;      // wx + 1
        int dd = d + (row / 3) - 1;
        int hh = h + (row % 3) - 1;
        int wx = wcol - 1;
        float v = 0.f;
        if ((unsigned)dd < 32u && (unsigned)hh < 32u && (unsigned)wx < 32u)
            v = xb[(size_t)ci * DHW + ((dd * 32 + hh) * 32 + wx)];
        x_lds[(row * 34 + wcol) * XSTR + ci] = (bf16)v;
    }

    const int lane  = tid & 31;
    const int wave  = tid >> 5;
    const int co_b  = wave * 16;
    const int lrow  = lane & 15;       // M (A) / N (B) index within tile
    const int koff  = (lane >> 4) * 8; // K sub-offset per half-wave

    v8f acc0 = {};
    v8f acc1 = {};

    for (int tap = 0; tap < NTAP; ++tap) {
        // Double buffer: issue tap+1's tile, then wait only for tap's tile.
        // 8 async ops/thread/tile, in-order completion => asynccnt<=8 means
        // tile[tap] has fully landed while tile[tap+1] is still in flight.
        if (tap + 1 < NTAP) {
            stage_a_tile(wsrc + (size_t)(tap + 1) * COUT * CIN,
                         a_lds[(tap + 1) & 1], tid);
#if USE_ASYNC
            asm volatile("s_wait_asynccnt 0x8" ::: "memory");
#endif
        } else {
#if USE_ASYNC
            asm volatile("s_wait_asynccnt 0x0" ::: "memory");
#endif
        }
        __syncthreads();   // tile[tap] + (first iter) x_lds visible to all

        const bf16* at = a_lds[tap & 1];
        const int kd = tap / 9, kh = (tap / 3) % 3, kw = tap % 3;
        const int rowbase = (kd * 3 + kh) * 34 + kw;

#pragma unroll
        for (int kc = 0; kc < 4; ++kc) {   // K = 128 ci in chunks of 32
            const int col = kc * 32 + koff;
            v16bf af = load_frag(&at[(co_b + lrow) * ASTR + col]);
            v16bf b0 = load_frag(&x_lds[(rowbase + lrow) * XSTR + col]);
            v16bf b1 = load_frag(&x_lds[(rowbase + 16 + lrow) * XSTR + col]);
            acc0 = __builtin_amdgcn_wmma_f32_16x16x32_bf16(
                       false, af, false, b0, (short)0, acc0, false, false);
            acc1 = __builtin_amdgcn_wmma_f32_16x16x32_bf16(
                       false, af, false, b1, (short)0, acc1, false, false);
        }
        __syncthreads();   // all waves done reading buf[tap&1] before reuse
    }

    // ---- Epilogue: noise + bias + leaky(0.2) * sqrt(2), clamp +-256 ----
    const float ns     = nstr[0];
    const float gain   = 1.4142135623730951f;
    const int   nbase  = (d * 32 + h) * 32;
    const float noise0 = noise[nbase + lrow] * ns;
    const float noise1 = noise[nbase + 16 + lrow] * ns;
    float* ob = out + (size_t)b * COUT * DHW + nbase;
    const int mb = co_b + (lane >> 4) * 8;   // C layout: lanes>=16 -> M = r+8

#pragma unroll
    for (int r = 0; r < 8; ++r) {
        int   m  = mb + r;
        float bv = bias[m];
        float y0 = acc0[r] + noise0 + bv;
        y0 = (y0 >= 0.f ? y0 : 0.2f * y0) * gain;
        y0 = fminf(fmaxf(y0, -256.f), 256.f);
        ob[(size_t)m * DHW + lrow] = y0;
        float y1 = acc1[r] + noise1 + bv;
        y1 = (y1 >= 0.f ? y1 : 0.2f * y1) * gain;
        y1 = fminf(fmaxf(y1, -256.f), 256.f);
        ob[(size_t)m * DHW + 16 + lrow] = y1;
    }
}

// ---------------------------------------------------------------------------
extern "C" void kernel_launch(void* const* d_in, const int* in_sizes, int n_in,
                              void* d_out, int out_size, void* d_ws,
                              size_t ws_size, hipStream_t stream) {
    (void)in_sizes; (void)n_in; (void)out_size; (void)ws_size;

    const float* x      = (const float*)d_in[0];
    const float* w      = (const float*)d_in[1];
    const float* aw     = (const float*)d_in[2];
    const float* ab     = (const float*)d_in[3];
    const float* weight = (const float*)d_in[4];
    const float* noise  = (const float*)d_in[5];
    const float* nstr   = (const float*)d_in[6];
    const float* bias   = (const float*)d_in[7];
    float*       out    = (float*)d_out;

    // workspace: styles (2KB) @0, bf16 modulated weights (3.46MB) @4096
    float* styles = (float*)d_ws;
    bf16*  wmod   = (bf16*)((char*)d_ws + 4096);

    styles_kernel  <<<2,    256, 0, stream>>>(w, aw, ab, styles);
    modulate_kernel<<<512,  128, 0, stream>>>(weight, styles, wmod);
    conv_kernel    <<<4096, 256, 0, stream>>>(x, wmod, noise, nstr, bias, out);
}
